// Encoder_conv_55035710931039
// MI455X (gfx1250) — matrix operand order repaired and checked by reading, not compiled, not run on hardware
//
#include <hip/hip_runtime.h>
#include <hip/hip_bf16.h>

// ---------------------------------------------------------------------------
// VQ-VAE encoder forward for MI455X (gfx1250, wave32, WMMA).
// Pipeline: conv1 -> conv2 -> conv3 -> res0 -> res1 -> [2x bf16-WMMA GEMM]
//           -> VQ (in place over mu) -> deterministic loss reduction.
// ---------------------------------------------------------------------------

typedef __attribute__((ext_vector_type(16))) __bf16 v16bf;
typedef __attribute__((ext_vector_type(8)))  float  v8f;

#define NB   16
#define NH   128
#define NW   256
#define C1   32
#define CC   64
#define W1   128
#define W2   64
#define HID  4096
#define GM   2048          // B*H rows of hf
#define NEMB 512
#define TOT1 (NB*C1*NH*W1) // 8388608
#define TOT2 (NB*CC*NH*W2) // 8388608

// ---------------- conv1: (16,1,128,256) -> (16,32,128,128), k=4,s=2,p=1 ----
__global__ __launch_bounds__(256) void k_conv1(const float* __restrict__ x,
                                               const float* __restrict__ w,
                                               const float* __restrict__ b,
                                               float* __restrict__ out) {
  int t = blockIdx.x * 256 + threadIdx.x;
  int ow = t & 127, hh = (t >> 7) & 127, oc = (t >> 14) & 31, bb = t >> 19;
  const float* xr = x + (bb * 128 + hh) * 256;
  const float* wr = w + oc * 4;
  float acc = b[oc];
  int iw0 = 2 * ow - 1;
#pragma unroll
  for (int k = 0; k < 4; ++k) {
    int iw = iw0 + k;
    float xv = (iw >= 0 && iw < 256) ? xr[iw] : 0.0f;
    acc = fmaf(xv, wr[k], acc);
  }
  out[t] = fmaxf(acc, 0.0f);
}

// ---------------- conv2: (16,32,128,128) -> (16,64,128,64), k=4,s=2,p=1 ----
__global__ __launch_bounds__(256) void k_conv2(const float* __restrict__ in,
                                               const float* __restrict__ w,
                                               const float* __restrict__ b,
                                               float* __restrict__ out) {
  int t = blockIdx.x * 256 + threadIdx.x;
  int ow = t & 63, hh = (t >> 6) & 127, oc = (t >> 13) & 63, bb = t >> 19;
  float acc = b[oc];
  int iw0 = 2 * ow - 1;
#pragma unroll 4
  for (int ic = 0; ic < 32; ++ic) {
    const float* ir = in + ((bb * 32 + ic) * 128 + hh) * 128;
    const float* wr = w + oc * 128 + ic * 4;
#pragma unroll
    for (int k = 0; k < 4; ++k) {
      int iw = iw0 + k;
      float xv = (iw >= 0 && iw < 128) ? ir[iw] : 0.0f;
      acc = fmaf(xv, wr[k], acc);
    }
  }
  out[t] = fmaxf(acc, 0.0f);
}

// ---------------- conv3: (16,64,128,64) -> same, k=3,s=1,p=1 ---------------
__global__ __launch_bounds__(256) void k_conv3(const float* __restrict__ in,
                                               const float* __restrict__ w,
                                               const float* __restrict__ b,
                                               float* __restrict__ out) {
  int t = blockIdx.x * 256 + threadIdx.x;
  int ow = t & 63, hh = (t >> 6) & 127, oc = (t >> 13) & 63, bb = t >> 19;
  float acc = b[oc];
  int iw0 = ow - 1;
#pragma unroll 4
  for (int ic = 0; ic < 64; ++ic) {
    const float* ir = in + ((bb * 64 + ic) * 128 + hh) * 64;
    const float* wr = w + oc * 192 + ic * 3;
#pragma unroll
    for (int k = 0; k < 3; ++k) {
      int iw = iw0 + k;
      float xv = (iw >= 0 && iw < 64) ? ir[iw] : 0.0f;
      acc = fmaf(xv, wr[k], acc);
    }
  }
  out[t] = fmaxf(acc, 0.0f);
}

// ---------------- fused residual block: h + W2*relu(W1*relu(h)) ------------
// One workgroup per (b,hh) row; 256-channel intermediate lives in LDS.
__global__ __launch_bounds__(256) void k_res(const float* __restrict__ hin,
                                             const float* __restrict__ w1,
                                             const float* __restrict__ w2,
                                             float* __restrict__ out) {
  __shared__ float s_h[64][66];   // relu(h) with halo
  __shared__ float s_y1[64][64];  // one 64-channel slab of relu(conv1)
  int bh = blockIdx.x;
  int bb = bh >> 7, hh = bh & 127;
  const float* base = hin + (size_t)bb * 524288 + hh * 64; // + ch*8192 + w
  int tid = threadIdx.x;

  for (int i = tid; i < 64 * 66; i += 256) {
    int ch = i / 66, wi = i % 66, w = wi - 1;
    float v = 0.0f;
    if (w >= 0 && w < 64) v = fmaxf(base[ch * 8192 + w], 0.0f);
    s_h[ch][wi] = v;
  }
  float acc[16];
#pragma unroll
  for (int r = 0; r < 16; ++r) acc[r] = 0.0f;
  __syncthreads();

  for (int g = 0; g < 4; ++g) {
    // phase A: 64 channels of conv(64->256, k=3) for this group
    for (int i = tid; i < 4096; i += 256) {
      int c4 = i >> 6, wp = i & 63;
      const float* wr = w1 + (g * 64 + c4) * 192;
      float v = 0.0f;
#pragma unroll 8
      for (int ic = 0; ic < 64; ++ic) {
        v = fmaf(s_h[ic][wp],     wr[ic * 3],     v);
        v = fmaf(s_h[ic][wp + 1], wr[ic * 3 + 1], v);
        v = fmaf(s_h[ic][wp + 2], wr[ic * 3 + 2], v);
      }
      s_y1[c4][wp] = fmaxf(v, 0.0f);
    }
    __syncthreads();
    // phase B: accumulate 1x1 conv (256->64) over this 64-channel slab
#pragma unroll
    for (int r = 0; r < 16; ++r) {
      int i = tid + 256 * r;
      int oc = i >> 6, wp = i & 63;
      const float* w2r = w2 + oc * 256 + g * 64;
      float v = acc[r];
#pragma unroll 8
      for (int j = 0; j < 64; ++j) v = fmaf(s_y1[j][wp], w2r[j], v);
      acc[r] = v;
    }
    __syncthreads();
  }
#pragma unroll
  for (int r = 0; r < 16; ++r) {
    int i = tid + 256 * r;
    int oc = i >> 6, wp = i & 63;
    out[(size_t)bb * 524288 + oc * 8192 + hh * 64 + wp] = base[oc * 8192 + wp] + acc[r];
  }
}

// ---------------- bf16 WMMA GEMM: out[m,n] = sum_k relu(A[m,k])*W[n,k]+b[n]
// M=2048, N=4096, K=4096.  WG tile 128x128 (8 waves, each 32x64 = 2x4 WMMA).
template <bool RELU>
__device__ inline v16bf load_frag(const float* __restrict__ p0,
                                  const float* __restrict__ p1) {
  const float4* a = reinterpret_cast<const float4*>(p0);
  const float4* b = reinterpret_cast<const float4*>(p1);
  float4 q0 = a[0], q1 = a[1], q2 = b[0], q3 = b[1];
  v16bf v;
  float f;
#define CVT(i, val)                          \
  f = (val);                                 \
  if (RELU) f = fmaxf(f, 0.0f);              \
  v[i] = (__bf16)f;
  CVT(0, q0.x)  CVT(1, q0.y)  CVT(2, q0.z)  CVT(3, q0.w)
  CVT(4, q1.x)  CVT(5, q1.y)  CVT(6, q1.z)  CVT(7, q1.w)
  CVT(8, q2.x)  CVT(9, q2.y)  CVT(10, q2.z) CVT(11, q2.w)
  CVT(12, q3.x) CVT(13, q3.y) CVT(14, q3.z) CVT(15, q3.w)
#undef CVT
  return v;
}

__global__ __launch_bounds__(256) void k_gemm_bf16_wmma(
    const float* __restrict__ A, const float* __restrict__ Wm,
    const float* __restrict__ bias, float* __restrict__ out) {
  const int tid  = threadIdx.x;
  const int lane = tid & 31;
  const int wv   = tid >> 5;          // 0..7
  const int half = lane >> 4;         // 0/1
  const int l15  = lane & 15;
  const int wgM  = blockIdx.y * 128 + (wv >> 1) * 32;
  const int wgN  = blockIdx.x * 128 + (wv & 1) * 64;

  v8f acc[2][4];
  const v8f vzero = {0.f, 0.f, 0.f, 0.f, 0.f, 0.f, 0.f, 0.f};
#pragma unroll
  for (int mt = 0; mt < 2; ++mt)
#pragma unroll
    for (int nt = 0; nt < 4; ++nt) acc[mt][nt] = vzero;

  const float* arow[2];
#pragma unroll
  for (int mt = 0; mt < 2; ++mt)
    arow[mt] = A + (size_t)(wgM + mt * 16 + l15) * HID;
  const float* brow[4];
#pragma unroll
  for (int nt = 0; nt < 4; ++nt)
    brow[nt] = Wm + (size_t)(wgN + nt * 16 + l15) * HID;

  for (int kb = 0; kb < HID; kb += 32) {
    // A frag (16x32 bf16): two 8-elem K runs at kb+8*half and kb+16+8*half
    const int ar0 = kb + half * 8;
    const int ar1 = kb + 16 + half * 8;
    // B frag (32x16 bf16): two 8-elem K runs at kb+8*half and kb+16+8*half (lane = N column)
    const int br0 = kb + half * 8;

    v16bf af[2], bf[4];
#pragma unroll
    for (int mt = 0; mt < 2; ++mt)
      af[mt] = load_frag<true>(arow[mt] + ar0, arow[mt] + ar1);
#pragma unroll
    for (int nt = 0; nt < 4; ++nt)
      bf[nt] = load_frag<false>(brow[nt] + br0, brow[nt] + br0 + 16);

    if (kb + 32 < HID) {                       // global_prefetch_b8 path
      __builtin_prefetch(arow[0] + kb + 32, 0, 1);
      __builtin_prefetch(brow[0] + kb + 32, 0, 1);
      __builtin_prefetch(brow[2] + kb + 32, 0, 1);
    }

#pragma unroll
    for (int mt = 0; mt < 2; ++mt)
#pragma unroll
      for (int nt = 0; nt < 4; ++nt)
        acc[mt][nt] = __builtin_amdgcn_wmma_f32_16x16x32_bf16(
            false, af[mt], false, bf[nt], (short)0, acc[mt][nt], false, false);
  }

  // D layout: M = r + 8*half, N = lane&15
#pragma unroll
  for (int mt = 0; mt < 2; ++mt)
#pragma unroll
    for (int nt = 0; nt < 4; ++nt) {
      const int n = wgN + nt * 16 + l15;
      const float bb = bias[n];
#pragma unroll
      for (int r = 0; r < 8; ++r) {
        const int m = wgM + mt * 16 + half * 8 + r;
        out[(size_t)m * HID + n] = acc[mt][nt][r] + bb;
      }
    }
}

// ---------------- VQ: argmin over 512 codes, quantize in place -------------
__global__ __launch_bounds__(256) void k_vq(float* __restrict__ zq,
                                            const float* __restrict__ emb,
                                            float* __restrict__ partial) {
  __shared__ float s_emb[8192];  // 128 codes x 64 dims (32 KB)
  __shared__ float s_red[256];
  int tid = threadIdx.x;
  int p = blockIdx.x * 256 + tid;                 // 0..131071
  int bb = p >> 13, rem = p & 8191, hh = rem >> 6, w = rem & 63;
  float* zbase = zq + (size_t)bb * 524288 + hh * 64 + w;  // + cc*8192

  float z[64];
#pragma unroll
  for (int cc = 0; cc < 64; ++cc) z[cc] = zbase[cc * 8192];

  float bestd = 3.4e38f;
  int bestidx = 0;
  for (int ch = 0; ch < 4; ++ch) {
    const float4* esrc = reinterpret_cast<const float4*>(emb + ch * 8192);
    float4* edst = reinterpret_cast<float4*>(s_emb);
    for (int i = tid; i < 2048; i += 256) edst[i] = esrc[i];
    __syncthreads();
    for (int e = 0; e < 128; ++e) {
      const float4* ev4 = reinterpret_cast<const float4*>(&s_emb[e << 6]);
      float d = 0.0f;
#pragma unroll
      for (int c4 = 0; c4 < 16; ++c4) {
        float4 e4 = ev4[c4];
        float d0 = z[4 * c4]     - e4.x;
        float d1 = z[4 * c4 + 1] - e4.y;
        float d2 = z[4 * c4 + 2] - e4.z;
        float d3 = z[4 * c4 + 3] - e4.w;
        d = fmaf(d0, d0, d); d = fmaf(d1, d1, d);
        d = fmaf(d2, d2, d); d = fmaf(d3, d3, d);
      }
      if (d < bestd) { bestd = d; bestidx = ch * 128 + e; }
    }
    __syncthreads();
  }

  const float* ev = emb + bestidx * 64;
  float err = 0.0f;
#pragma unroll
  for (int cc = 0; cc < 64; ++cc) {
    float q = ev[cc];
    float df = q - z[cc];
    err = fmaf(df, df, err);
    zbase[cc * 8192] = q;          // quantized, same flat layout as mu
  }

  s_red[tid] = err;
  __syncthreads();
#pragma unroll
  for (int s = 128; s > 0; s >>= 1) {
    if (tid < s) s_red[tid] += s_red[tid + s];
    __syncthreads();
  }
  if (tid == 0) partial[blockIdx.x] = s_red[0];
}

// ---------------- deterministic final loss reduction -----------------------
__global__ __launch_bounds__(256) void k_loss(const float* __restrict__ partial,
                                              float* __restrict__ loss) {
  __shared__ float s[256];
  int tid = threadIdx.x;
  s[tid] = partial[tid] + partial[tid + 256];
  __syncthreads();
#pragma unroll
  for (int st = 128; st > 0; st >>= 1) {
    if (tid < st) s[tid] += s[tid + st];
    __syncthreads();
  }
  // loss = q_loss + e_loss = 2 * mean((q - lat)^2) over 8388608 elements
  if (tid == 0) loss[0] = s[0] * (2.0f / 8388608.0f);
}

// ---------------------------------------------------------------------------
extern "C" void kernel_launch(void* const* d_in, const int* in_sizes, int n_in,
                              void* d_out, int out_size, void* d_ws, size_t ws_size,
                              hipStream_t stream) {
  const float* x     = (const float*)d_in[0];
  const float* in_w  = (const float*)d_in[1];
  const float* in_b  = (const float*)d_in[2];
  const float* hd_w  = (const float*)d_in[3];
  const float* hd_b  = (const float*)d_in[4];
  const float* hd2_w = (const float*)d_in[5];
  const float* hd2_b = (const float*)d_in[6];
  const float* r0w1  = (const float*)d_in[7];
  const float* r0w2  = (const float*)d_in[8];
  const float* r1w1  = (const float*)d_in[9];
  const float* r1w2  = (const float*)d_in[10];
  const float* mu_w  = (const float*)d_in[11];
  const float* mu_b  = (const float*)d_in[12];
  const float* c_w   = (const float*)d_in[13];
  const float* c_b   = (const float*)d_in[14];
  const float* emb   = (const float*)d_in[15];

  float* out     = (float*)d_out;
  float* bufA    = (float*)d_ws;            // 33.5 MB
  float* bufB    = bufA + 8388608;          // 33.5 MB
  float* partial = bufB + 8388608;          // 2 KB

  float* mu_out = out;                      // quantized slot; VQ runs in place
  float* loss_p = out + 8388608;
  float* c_out  = out + 8388609;

  k_conv1<<<TOT1 / 256, 256, 0, stream>>>(x, in_w, in_b, bufA);
  k_conv2<<<TOT2 / 256, 256, 0, stream>>>(bufA, hd_w, hd_b, bufB);
  k_conv3<<<TOT2 / 256, 256, 0, stream>>>(bufB, hd2_w, hd2_b, bufA);
  k_res<<<2048, 256, 0, stream>>>(bufA, r0w1, r0w2, bufB);
  k_res<<<2048, 256, 0, stream>>>(bufB, r1w1, r1w2, bufA);

  dim3 gg(HID / 128, GM / 128);             // 32 x 16 workgroups
  k_gemm_bf16_wmma<<<gg, 256, 0, stream>>>(bufA, mu_w, mu_b, mu_out);
  k_gemm_bf16_wmma<<<gg, 256, 0, stream>>>(bufA, c_w, c_b, c_out);

  k_vq<<<512, 256, 0, stream>>>(mu_out, emb, partial);
  k_loss<<<1, 256, 0, stream>>>(partial, loss_p);
}
